// CLAPLoss_46136538694118
// MI455X (gfx1250) — matrix operand order repaired and checked by reading, not compiled, not run on hardware
//
#include <hip/hip_runtime.h>
#include <hip/hip_bf16.h>
#include <math.h>

typedef __attribute__((ext_vector_type(16))) _Float16 v16h;
typedef __attribute__((ext_vector_type(8)))  _Float16 v8h;
typedef __attribute__((ext_vector_type(8)))  float    v8f;

#define B 256
#define F 128
#define Q 4096
#define E 128
#define TQ 64
#define NQT (Q / TQ)
#define NB_LDK (F + 8)   // padded K-stride of transposed B tile (f16 elems, mult of 8)
#define INV_TAU 5.0f

// d_out layout (flat float): moco_loss | logits(B x 4097) | easy_loss |
// easy_mask(B x Q) | hard_loss | hard_mask(B x Q)
#define O_LOGITS 1
#define O_EASYL  1048833
#define O_EASYM  1048834
#define O_HARDL  2097410
#define O_HARDM  2097411

#if defined(__gfx1250__) && __has_builtin(__builtin_amdgcn_global_load_async_to_lds_b128)
#define HAS_ASYNC 1
#else
#define HAS_ASYNC 0
#endif

#if HAS_ASYNC
typedef __attribute__((__vector_size__(4 * sizeof(int)))) int v4i_;
__device__ __forceinline__ void async_g2l_b128(const void* g, void* l) {
  __builtin_amdgcn_global_load_async_to_lds_b128((v4i_*)g, (v4i_*)l, 0, 0);
}
#if __has_builtin(__builtin_amdgcn_s_wait_asynccnt)
#define WAIT_ASYNC0() __builtin_amdgcn_s_wait_asynccnt(0)
#else
#define WAIT_ASYNC0() asm volatile("s_wait_asynccnt 0" ::: "memory")
#endif
#endif

// ---- WMMA fragment loaders (v_wmma_f32_16x16x32_f16 operand layouts) ----
// A 16x32 f16 per lane = two contiguous 8-halfword runs: K = kb+hl*8+{0..7} and +16.
__device__ __forceinline__ v16h load_afrag(const _Float16* t, int ldt, int rowBase,
                                           int kbase, int hl, int lc) {
  const _Float16* p = t + (rowBase + lc) * ldt + kbase + hl * 8;
  v8h lo = *(const v8h*)(p);
  v8h hi = *(const v8h*)(p + 16);
  return __builtin_shufflevector(lo, hi, 0, 1, 2, 3, 4, 5, 6, 7,
                                 8, 9, 10, 11, 12, 13, 14, 15);
}
// B 32x16 f16: lane (hl,lc) needs column colBase+lc, K = kbase+hl*8+{0..7} and +16:
// contiguous in the transposed LDS tile nBT[col][k].
__device__ __forceinline__ v16h load_bfrag(const _Float16* nT, int col, int kbase, int hl) {
  const _Float16* p = nT + col * NB_LDK + kbase + hl * 8;
  v8h lo = *(const v8h*)(p);
  v8h hi = *(const v8h*)(p + 16);
  return __builtin_shufflevector(lo, hi, 0, 1, 2, 3, 4, 5, 6, 7,
                                 8, 9, 10, 11, 12, 13, 14, 15);
}

// ===================== fused per-e similarity kernel =====================
// grid = E, block = 256 (8 waves). Per e:
//   t = l2norm(mask_t?0:teacher), n = l2norm(mask_n?0:negatives)
//   sim = t@n / TAU  (WMMA f16->f32, streamed 64-col tiles, never stored to HBM)
//   pass A: online row (axis-Q) softmax stats with score_mask,
//           full column (axis-B) softmax + entropy with boot_mask -> unc_acc
//   pass B: recompute sim, scatter normalized probs into scores_acc (atomics)
// Tile fetch is software-pipelined: tile t+1 staged (async-to-LDS when
// available, ASYNCcnt-tracked) before the WMMA burst of tile t.
__global__ __launch_bounds__(256, 1)
void sim_kernel(const float* __restrict__ teacher, const float* __restrict__ negatives,
                const unsigned char* __restrict__ mask_t, const unsigned char* __restrict__ mask_n,
                const unsigned char* __restrict__ smask, const unsigned char* __restrict__ bmask,
                float* __restrict__ scores_acc, float* __restrict__ unc_acc) {
  __shared__ __align__(32) unsigned char smem[65536];
  _Float16* tA     = (_Float16*)smem;            // prologue only (64KB, aliased)
  float*    rawF   = (float*)smem;               // raw f32 tile [128][64] = 32768B
  _Float16* nBT    = (_Float16*)(smem + 32768);  // 64 x NB_LDK f16 = 17408B
  float* colred    = (float*)(smem + 50176);     // 8 x 64
  float* colred2   = (float*)(smem + 52224);     // 8 x 64
  float* colmax    = (float*)(smem + 54272);     // 64
  float* colsum    = (float*)(smem + 54528);     // 64
  float* colcnt    = (float*)(smem + 54784);     // 64
  float* cnorm     = (float*)(smem + 55040);     // 64
  float* rowm      = (float*)(smem + 55296);     // 256 (persists after prologue)
  float* rows_     = (float*)(smem + 56320);     // 256

  const int tid = threadIdx.x;
  const int e   = blockIdx.x;
  const int wave = tid >> 5, lane = tid & 31, hl = lane >> 4, lc = lane & 15;
  const int eBQ = e * B * Q;
  const int eFQ = e * F * Q;

  // ---- prologue: normalized masked teacher -> tA, then A frags to regs ----
  {
    int r = tid;
    const float* tr = teacher + r * F;
    const unsigned char* mr = mask_t + (e * B + r) * F;
    float ss = 0.f;
    for (int f = 0; f < F; ++f) { float v = mr[f] ? 0.f : tr[f]; ss += v * v; }
    float inv = 1.f / fmaxf(sqrtf(ss), 1e-12f);
    for (int c = 0; c < 16; ++c) {
      v8h t8;
#pragma unroll
      for (int j = 0; j < 8; ++j) {
        int f = c * 8 + j;
        float v = mr[f] ? 0.f : tr[f];
        t8[j] = (_Float16)(v * inv);
      }
      *(v8h*)(tA + r * F + c * 8) = t8;
    }
  }
  __syncthreads();
  v16h afr[2][4];
#pragma unroll
  for (int m = 0; m < 2; ++m)
#pragma unroll
    for (int kk = 0; kk < 4; ++kk)
      afr[m][kk] = load_afrag(tA, F, (wave * 2 + m) * 16, kk * 32, hl, lc);
  __syncthreads();
  rowm[tid] = -INFINITY; rows_[tid] = 0.f;
  __syncthreads();

  // ---- stage tile q0: issue raw loads (async) + mask bits to a register ----
  auto stageIssue = [&](int q0) -> unsigned {
    int col = tid & 63, seg = tid >> 6;
#if HAS_ASYNC
    {
      int cb = (lane & 15) * 4;           // 4-float column block
      int rb = wave * 16 + (lane >> 4);   // row parity within wave's 16 rows
#pragma unroll
      for (int i = 0; i < 8; ++i) {
        int f = rb + i * 2;
        async_g2l_b128(negatives + (size_t)f * Q + q0 + cb, rawF + f * TQ + cb);
      }
    }
#endif
    unsigned mb = 0u;
#pragma unroll 8
    for (int j = 0; j < 32; ++j)
      if (mask_n[eFQ + (seg * 32 + j) * Q + q0 + col]) mb |= (1u << j);
    return mb;
  };

  // ---- finish tile: column l2-norm + pack transposed f16 B tile ----
  auto finishN = [&](int q0, unsigned mb) {
    int col = tid & 63, seg = tid >> 6;
#if HAS_ASYNC
    WAIT_ASYNC0();
    __syncthreads();          // all waves' async raw data landed
#endif
    float vals[32];
    float ss = 0.f;
#pragma unroll 4
    for (int j = 0; j < 32; ++j) {
      float v;
#if HAS_ASYNC
      v = ((mb >> j) & 1u) ? 0.f : rawF[(seg * 32 + j) * TQ + col];
#else
      v = ((mb >> j) & 1u) ? 0.f : negatives[(size_t)(seg * 32 + j) * Q + q0 + col];
#endif
      vals[j] = v; ss += v * v;
    }
    colred[seg * TQ + col] = ss;
    __syncthreads();
    if (tid < TQ) {
      float s = colred[tid] + colred[TQ + tid] + colred[2 * TQ + tid] + colred[3 * TQ + tid];
      cnorm[tid] = 1.f / fmaxf(sqrtf(s), 1e-12f);
    }
    __syncthreads();
    float inv = cnorm[col];
    _Float16* dst = nBT + col * NB_LDK + seg * 32;
#pragma unroll
    for (int c = 0; c < 4; ++c) {
      v8h t8;
#pragma unroll
      for (int j = 0; j < 8; ++j) t8[j] = (_Float16)(vals[c * 8 + j] * inv);
      *(v8h*)(dst + c * 8) = t8;
    }
    __syncthreads();          // raw fully consumed; nBT visible
  };

  auto doGemm = [&](v8f (&acc)[2][4]) {
#pragma unroll
    for (int m = 0; m < 2; ++m)
#pragma unroll
      for (int n = 0; n < 4; ++n)
#pragma unroll
        for (int r = 0; r < 8; ++r) acc[m][n][r] = 0.f;
#pragma unroll
    for (int kk = 0; kk < 4; ++kk)
#pragma unroll
      for (int n = 0; n < 4; ++n) {
        v16h bf = load_bfrag(nBT, n * 16 + lc, kk * 32, hl);
        acc[0][n] = __builtin_amdgcn_wmma_f32_16x16x32_f16(false, afr[0][kk], false, bf,
                                                           (short)0, acc[0][n], false, false);
        acc[1][n] = __builtin_amdgcn_wmma_f32_16x16x32_f16(false, afr[1][kk], false, bf,
                                                           (short)0, acc[1][n], false, false);
      }
#pragma unroll
    for (int m = 0; m < 2; ++m)
#pragma unroll
      for (int n = 0; n < 4; ++n)
#pragma unroll
        for (int r = 0; r < 8; ++r) acc[m][n][r] *= INV_TAU;
  };

  // =============================== pass A ===============================
  unsigned mb = stageIssue(0);
  for (int qt = 0; qt < NQT; ++qt) {
    int q0 = qt * TQ;
    finishN(q0, mb);
    if (qt + 1 < NQT) mb = stageIssue(q0 + TQ);   // overlap with WMMA below
    v8f acc[2][4];
    doGemm(acc);

    unsigned long long sb = 0ull, bb = 0ull;
#pragma unroll
    for (int m = 0; m < 2; ++m)
#pragma unroll
      for (int n = 0; n < 4; ++n)
#pragma unroll
        for (int r = 0; r < 8; ++r) {
          int row = (wave * 2 + m) * 16 + hl * 8 + r;
          int col = q0 + n * 16 + lc;
          int gi  = eBQ + row * Q + col;
          int bit = (m * 4 + n) * 8 + r;
          if (smask[gi]) sb |= (1ull << bit);
          if (bmask[gi]) bb |= (1ull << bit);
        }

    // row-wise online softmax stats (score_mask, axis Q); rows wave-exclusive
#pragma unroll
    for (int m = 0; m < 2; ++m)
#pragma unroll
      for (int r = 0; r < 8; ++r) {
        float vmax = -INFINITY;
#pragma unroll
        for (int n = 0; n < 4; ++n) {
          int bit = (m * 4 + n) * 8 + r;
          if ((sb >> bit) & 1ull) vmax = fmaxf(vmax, acc[m][n][r]);
        }
        for (int off = 1; off < 16; off <<= 1) vmax = fmaxf(vmax, __shfl_xor(vmax, off, 32));
        float ts = 0.f;
        if (vmax > -INFINITY) {
#pragma unroll
          for (int n = 0; n < 4; ++n) {
            int bit = (m * 4 + n) * 8 + r;
            if ((sb >> bit) & 1ull) ts += __expf(acc[m][n][r] - vmax);
          }
        }
        for (int off = 1; off < 16; off <<= 1) ts += __shfl_xor(ts, off, 32);
        if (lc == 0 && vmax > -INFINITY) {
          int row = (wave * 2 + m) * 16 + hl * 8 + r;
          float om = rowm[row];
          float nm = fmaxf(om, vmax);
          rows_[row] = rows_[row] * __expf(om - nm) + ts * __expf(vmax - nm);
          rowm[row]  = nm;
        }
      }

    // column-wise (boot_mask, axis B): full B present -> single pass
#pragma unroll
    for (int n = 0; n < 4; ++n) {
      float cm = -INFINITY;
#pragma unroll
      for (int m = 0; m < 2; ++m)
#pragma unroll
        for (int r = 0; r < 8; ++r) {
          int bit = (m * 4 + n) * 8 + r;
          if ((bb >> bit) & 1ull) cm = fmaxf(cm, acc[m][n][r]);
        }
      cm = fmaxf(cm, __shfl_xor(cm, 16, 32));
      if (hl == 0) colred[wave * TQ + n * 16 + lc] = cm;
    }
    __syncthreads();
    if (tid < TQ) {
      float m0 = colred[tid];
      for (int w = 1; w < 8; ++w) m0 = fmaxf(m0, colred[w * TQ + tid]);
      colmax[tid] = m0;
    }
    __syncthreads();
#pragma unroll
    for (int n = 0; n < 4; ++n) {
      float cmx = colmax[n * 16 + lc];
      float ps = 0.f, pc = 0.f;
#pragma unroll
      for (int m = 0; m < 2; ++m)
#pragma unroll
        for (int r = 0; r < 8; ++r) {
          int bit = (m * 4 + n) * 8 + r;
          if ((bb >> bit) & 1ull) { ps += __expf(acc[m][n][r] - cmx); pc += 1.f; }
        }
      ps += __shfl_xor(ps, 16, 32);
      pc += __shfl_xor(pc, 16, 32);
      if (hl == 0) { colred[wave * TQ + n * 16 + lc] = ps; colred2[wave * TQ + n * 16 + lc] = pc; }
    }
    __syncthreads();
    if (tid < TQ) {
      float s = 0.f, c = 0.f;
      for (int w = 0; w < 8; ++w) { s += colred[w * TQ + tid]; c += colred2[w * TQ + tid]; }
      colsum[tid] = s; colcnt[tid] = c;
    }
    __syncthreads();
    const float C0 = 1.6118095651e-6f;  // -(1e-7)*ln(1e-7), masked-out entropy term
#pragma unroll
    for (int n = 0; n < 4; ++n) {
      float cmx = colmax[n * 16 + lc];
      float S = fmaxf(colsum[n * 16 + lc], 1e-30f);
      float pe = 0.f;
#pragma unroll
      for (int m = 0; m < 2; ++m)
#pragma unroll
        for (int r = 0; r < 8; ++r) {
          int bit = (m * 4 + n) * 8 + r;
          if ((bb >> bit) & 1ull) {
            float p = __expf(acc[m][n][r] - cmx) / S + 1e-7f;
            pe += -(p * __logf(p));
          } else pe += C0;
        }
      pe += __shfl_xor(pe, 16, 32);
      if (hl == 0) colred[wave * TQ + n * 16 + lc] = pe;
    }
    __syncthreads();
    if (tid < TQ) {
      float ent = 0.f;
      for (int w = 0; w < 8; ++w) ent += colred[w * TQ + tid];
      atomicAdd(&unc_acc[q0 + tid], ent / fmaxf(colcnt[tid], 1.f));
    }
    __syncthreads();
  }
  __syncthreads();

  // ========== pass B: recompute sim, emit normalized probs ==========
  mb = stageIssue(0);
  for (int qt = 0; qt < NQT; ++qt) {
    int q0 = qt * TQ;
    finishN(q0, mb);
    if (qt + 1 < NQT) mb = stageIssue(q0 + TQ);
    v8f acc[2][4];
    doGemm(acc);
#pragma unroll
    for (int m = 0; m < 2; ++m)
#pragma unroll
      for (int r = 0; r < 8; ++r) {
        int row = (wave * 2 + m) * 16 + hl * 8 + r;
        float mR = rowm[row];
        float sR = fmaxf(rows_[row], 1e-30f);
#pragma unroll
        for (int n = 0; n < 4; ++n) {
          int col = q0 + n * 16 + lc;
          if (smask[eBQ + row * Q + col]) {
            float p = __expf(acc[m][n][r] - mR) / sR;
            atomicAdd(&scores_acc[row * Q + col], p);
          }
        }
      }
    __syncthreads();
  }
}

// ===================== cnt_e = max(sum_e score_mask, 1) =====================
__global__ void cnt_kernel(const unsigned char* __restrict__ smask, float* __restrict__ cnt) {
  int i = blockIdx.x * 256 + threadIdx.x;
  int c = 0;
  for (int e = 0; e < E; ++e) c += smask[(size_t)e * B * Q + i];
  cnt[i] = fmaxf((float)c, 1.f);
}

__global__ void fin_scores_kernel(float* __restrict__ s, const float* __restrict__ cnt) {
  int i = blockIdx.x * 256 + threadIdx.x;
  s[i] = s[i] / cnt[i];
}

// ===================== moco logits: query @ negatives (WMMA) =====================
__global__ __launch_bounds__(256, 1)
void moco_gemm_kernel(const float* __restrict__ query, const float* __restrict__ negatives,
                      float* __restrict__ logits) {
  __shared__ __align__(32) unsigned char smem[65536];
  _Float16* tA  = (_Float16*)smem;
  _Float16* nBT = (_Float16*)smem;
  const int tid = threadIdx.x;
  const int q0 = blockIdx.x * TQ;
  const int wave = tid >> 5, lane = tid & 31, hl = lane >> 4, lc = lane & 15;
  {
    int r = tid;
    for (int c = 0; c < 16; ++c) {
      v8h t8;
#pragma unroll
      for (int j = 0; j < 8; ++j) t8[j] = (_Float16)query[r * F + c * 8 + j];
      *(v8h*)(tA + r * F + c * 8) = t8;
    }
  }
  __syncthreads();
  v16h afr[2][4];
#pragma unroll
  for (int m = 0; m < 2; ++m)
#pragma unroll
    for (int kk = 0; kk < 4; ++kk)
      afr[m][kk] = load_afrag(tA, F, (wave * 2 + m) * 16, kk * 32, hl, lc);
  __syncthreads();
  {
    int col = tid & 63, seg = tid >> 6;
    _Float16* dst = nBT + col * NB_LDK + seg * 32;
#pragma unroll
    for (int c = 0; c < 4; ++c) {
      v8h t8;
#pragma unroll
      for (int j = 0; j < 8; ++j)
        t8[j] = (_Float16)negatives[(seg * 32 + c * 8 + j) * Q + q0 + col];
      *(v8h*)(dst + c * 8) = t8;
    }
  }
  __syncthreads();
  v8f acc[2][4];
#pragma unroll
  for (int m = 0; m < 2; ++m)
#pragma unroll
    for (int n = 0; n < 4; ++n)
#pragma unroll
      for (int r = 0; r < 8; ++r) acc[m][n][r] = 0.f;
#pragma unroll
  for (int kk = 0; kk < 4; ++kk)
#pragma unroll
    for (int n = 0; n < 4; ++n) {
      v16h bf = load_bfrag(nBT, n * 16 + lc, kk * 32, hl);
      acc[0][n] = __builtin_amdgcn_wmma_f32_16x16x32_f16(false, afr[0][kk], false, bf,
                                                         (short)0, acc[0][n], false, false);
      acc[1][n] = __builtin_amdgcn_wmma_f32_16x16x32_f16(false, afr[1][kk], false, bf,
                                                         (short)0, acc[1][n], false, false);
    }
#pragma unroll
  for (int m = 0; m < 2; ++m)
#pragma unroll
    for (int n = 0; n < 4; ++n)
#pragma unroll
      for (int r = 0; r < 8; ++r) {
        int row = (wave * 2 + m) * 16 + hl * 8 + r;
        int col = q0 + n * 16 + lc;
        logits[(size_t)row * (Q + 1) + 1 + col] = acc[m][n][r] * INV_TAU;
      }
}

__global__ void pos_kernel(const float* __restrict__ q, const float* __restrict__ k,
                           float* __restrict__ logits) {
  int b = threadIdx.x;
  float s = 0.f;
  for (int f = 0; f < F; ++f) s += q[b * F + f] * k[b * F + f];
  logits[(size_t)b * (Q + 1)] = s * INV_TAU;
}

// per-row logsumexp (full + neg-only) -> moco loss + lse_neg
__global__ __launch_bounds__(256, 1)
void moco_reduce_kernel(const float* __restrict__ logits, float* __restrict__ lse_neg,
                        float* __restrict__ moco_out) {
  __shared__ float red[256];
  __shared__ float bc[2];
  int b = blockIdx.x, tid = threadIdx.x;
  const float* lg = logits + (size_t)b * (Q + 1);
  float mf = -INFINITY, mn = -INFINITY;
  for (int i = tid; i < Q + 1; i += 256) {
    float v = lg[i]; mf = fmaxf(mf, v); if (i >= 1) mn = fmaxf(mn, v);
  }
  red[tid] = mf; __syncthreads();
  for (int s = 128; s > 0; s >>= 1) { if (tid < s) red[tid] = fmaxf(red[tid], red[tid + s]); __syncthreads(); }
  if (tid == 0) bc[0] = red[0];
  __syncthreads();
  red[tid] = mn; __syncthreads();
  for (int s = 128; s > 0; s >>= 1) { if (tid < s) red[tid] = fmaxf(red[tid], red[tid + s]); __syncthreads(); }
  if (tid == 0) bc[1] = red[0];
  __syncthreads();
  mf = bc[0]; mn = bc[1];
  float sf = 0.f, sn = 0.f;
  for (int i = tid; i < Q + 1; i += 256) {
    float v = lg[i]; sf += __expf(v - mf); if (i >= 1) sn += __expf(v - mn);
  }
  red[tid] = sf; __syncthreads();
  for (int s = 128; s > 0; s >>= 1) { if (tid < s) red[tid] += red[tid + s]; __syncthreads(); }
  if (tid == 0) bc[0] = red[0];
  __syncthreads();
  red[tid] = sn; __syncthreads();
  for (int s = 128; s > 0; s >>= 1) { if (tid < s) red[tid] += red[tid + s]; __syncthreads(); }
  if (tid == 0) {
    lse_neg[b] = mn + __logf(red[0]);
    atomicAdd(moco_out, (mf + __logf(bc[0]) - lg[0]) * (1.0f / B));
  }
}

// uncertainty finalize + median via counting selection (EASY_UB == HARD_LB == 0.5)
__global__ __launch_bounds__(256, 1)
void median_kernel(const float* __restrict__ unc_acc, float* __restrict__ unc,
                   float* __restrict__ sel) {
  __shared__ float u[Q];
  int tid = threadIdx.x;
  for (int q = tid; q < Q; q += 256) { float v = unc_acc[q] * (1.0f / E); u[q] = v; unc[q] = v; }
  __syncthreads();
  for (int i = tid; i < Q; i += 256) {
    float v = u[i]; int cl = 0, ce = 0;
    for (int j = 0; j < Q; ++j) { float x = u[j]; cl += (x < v); ce += (x == v); }
    if (cl <= 2047 && 2047 < cl + ce) sel[0] = v;  // order stat 2047 (unique value)
    if (cl <= 2048 && 2048 < cl + ce) sel[1] = v;  // order stat 2048
  }
}

// per-row masked argmax (top-1, lowest-index tie-break) -> losses + masks
__global__ __launch_bounds__(256, 1)
void pos_loss_kernel(const float* __restrict__ scores, const float* __restrict__ unc,
                     const float* __restrict__ sel, const float* __restrict__ logits,
                     const float* __restrict__ lse_neg, float* __restrict__ out) {
  __shared__ float sv[256];
  __shared__ int   si[256];
  int b = blockIdx.x, tid = threadIdx.x;
  float med = 0.5f * (sel[0] + sel[1]);
  const float* sr = scores + (size_t)b * Q;
  for (int pass = 0; pass < 2; ++pass) {
    float bv = -INFINITY; int bi = 0x7fffffff;
    for (int q = tid; q < Q; q += 256) {
      bool ok = (pass == 0) ? (unc[q] <= med) : (unc[q] >= med);
      float v = ok ? sr[q] : -INFINITY;
      if (v > bv || (v == bv && q < bi)) { bv = v; bi = q; }
    }
    sv[tid] = bv; si[tid] = bi; __syncthreads();
    for (int s = 128; s > 0; s >>= 1) {
      if (tid < s) {
        float v = sv[tid + s]; int i = si[tid + s];
        if (v > sv[tid] || (v == sv[tid] && i < si[tid])) { sv[tid] = v; si[tid] = i; }
      }
      __syncthreads();
    }
    if (tid == 0) {
      int idx = si[0];
      long mOff = (pass == 0) ? (long)O_EASYM : (long)O_HARDM;
      long lOff = (pass == 0) ? (long)O_EASYL : (long)O_HARDL;
      out[mOff + (size_t)b * Q + idx] = 1.0f;
      float lsm = logits[(size_t)b * (Q + 1) + 1 + idx] - lse_neg[b];
      atomicAdd(&out[lOff], -lsm * (1.0f / B));
    }
    __syncthreads();
  }
}

extern "C" void kernel_launch(void* const* d_in, const int* in_sizes, int n_in,
                              void* d_out, int out_size, void* d_ws, size_t ws_size,
                              hipStream_t stream) {
  (void)in_sizes; (void)n_in; (void)ws_size;
  const float* query            = (const float*)d_in[0];
  const float* key_             = (const float*)d_in[1];
  const float* teacher          = (const float*)d_in[2];
  const float* negatives        = (const float*)d_in[3];
  const unsigned char* mask_t   = (const unsigned char*)d_in[4];
  const unsigned char* mask_n   = (const unsigned char*)d_in[5];
  const unsigned char* smask    = (const unsigned char*)d_in[6];
  const unsigned char* bmask    = (const unsigned char*)d_in[7];
  float* out = (float*)d_out;

  float* ws        = (float*)d_ws;
  float* scores    = ws;                          // B*Q
  float* cnt       = scores + (size_t)B * Q;      // B*Q
  float* unc_acc   = cnt + (size_t)B * Q;         // Q
  float* unc       = unc_acc + Q;                 // Q
  float* lse_neg   = unc + Q;                     // B
  float* sel       = lse_neg + B;                 // 2

  (void)hipMemsetAsync(d_out, 0, (size_t)out_size * sizeof(float), stream);
  (void)hipMemsetAsync(scores, 0, (size_t)B * Q * sizeof(float), stream);
  (void)hipMemsetAsync(unc_acc, 0, (size_t)Q * sizeof(float), stream);

  cnt_kernel<<<(B * Q) / 256, 256, 0, stream>>>(smask, cnt);
  sim_kernel<<<E, 256, 0, stream>>>(teacher, negatives, mask_t, mask_n, smask, bmask,
                                    scores, unc_acc);
  fin_scores_kernel<<<(B * Q) / 256, 256, 0, stream>>>(scores, cnt);
  median_kernel<<<1, 256, 0, stream>>>(unc_acc, unc, sel);
  moco_gemm_kernel<<<Q / TQ, 256, 0, stream>>>(query, negatives, out + O_LOGITS);
  pos_kernel<<<1, 256, 0, stream>>>(query, key_, out + O_LOGITS);
  moco_reduce_kernel<<<B, 256, 0, stream>>>(out + O_LOGITS, lse_neg, out + 0);
  pos_loss_kernel<<<B, 256, 0, stream>>>(scores, unc, sel, out + O_LOGITS, lse_neg, out);
}
